// Attn_51127290691658
// MI455X (gfx1250) — matrix operand order repaired and checked by reading, not compiled, not run on hardware
//
#include <hip/hip_runtime.h>
#include <math.h>
#include <stdint.h>

// Shapes from the reference
#define S_LEN 1024
#define B_SZ  64
#define H_SZ  512

typedef __bf16  v16bf __attribute__((ext_vector_type(16)));
typedef __bf16  v8bf  __attribute__((ext_vector_type(8)));
typedef float   v8f   __attribute__((ext_vector_type(8)));

typedef unsigned int v4u __attribute__((ext_vector_type(4)));
typedef int          v8i __attribute__((ext_vector_type(8)));
typedef int          v4i __attribute__((ext_vector_type(4)));

// ---------------- Kernel 1: convert W2 (= W_attn[:, H:2H]) to bf16 [N][K] ----------------
__global__ void w2_convert_kernel(const float* __restrict__ W, __bf16* __restrict__ w2) {
    int i = blockIdx.x * blockDim.x + threadIdx.x;          // 512*512 elements
    int n = i >> 9;
    int k = i & 511;
    w2[i] = (__bf16)W[(size_t)n * 1024 + 512 + k];
}

// ---------------- Kernel 2: hterm[b][n] = b_attn[n] + sum_k hidden[b,k]*W_attn[n,k] ------
__global__ void hterm_kernel(const float* __restrict__ hidden,
                             const float* __restrict__ W,
                             const float* __restrict__ bias,
                             float* __restrict__ hterm) {
    int i = blockIdx.x * blockDim.x + threadIdx.x;          // 64*512 threads
    int b = i >> 9;
    int n = i & 511;
    const float* hrow = hidden + (size_t)b * 512;
    const float* wrow = W + (size_t)n * 1024;               // first H columns = hidden part
    float acc = bias[n];
#pragma unroll 4
    for (int k = 0; k < 512; ++k) acc = fmaf(hrow[k], wrow[k], acc);
    hterm[i] = acc;
}

// ---------------- TDM helper: 2D tile load, global -> padded LDS -------------------------
// Builds a Tensor DMA Descriptor (D#) per CDNA5 ISA and issues
// tensor_load_to_lds (clang-23 6-arg builtin). Groups 2/3 are zero (2D tile,
// tile_dim2 = 0 -> unused).
__device__ __forceinline__ void tdm_load_2d(unsigned lds_off, const void* gaddr,
                                            int flags_d0, int tile_d0, int tile_d1,
                                            int tensor_d0, int stride0) {
    unsigned long long ga = (unsigned long long)(uintptr_t)gaddr;
    v4u g0;
    g0[0] = 1u;                                             // count=1, user mode
    g0[1] = lds_off;                                        // lds_addr (bytes)
    g0[2] = (unsigned)ga;                                   // global_addr[31:0]
    g0[3] = (unsigned)((ga >> 32) & 0x1FFFFFFu) | (2u << 30); // global_addr[56:32], type=2

    v8i g1;
    g1[0] = flags_d0;                                       // data_size | pad_enable | pad_interval | pad_amount
    g1[1] = (tensor_d0 & 0xFFFF) << 16;                     // tensor_dim0[15:0]
    g1[2] = (tensor_d0 >> 16) & 0xFFFF;                     // tensor_dim0[31:16] | tensor_dim1[15:0] (=0, dim1=1<<20)
    g1[3] = ((1 << 20) >> 16) | (tile_d0 << 16);            // tensor_dim1[31:16]=16 | tile_dim0
    g1[4] = tile_d1;                                        // tile_dim1 | tile_dim2=0
    g1[5] = stride0;                                        // tensor_dim0_stride[31:0]
    g1[6] = 0;                                              // stride0[47:32]=0 | dim1_stride[15:0]=0
    g1[7] = 0;                                              // dim1_stride[47:16]=0

    v4i z4 = (v4i){0, 0, 0, 0};
    v8i z8 = (v8i){0, 0, 0, 0, 0, 0, 0, 0};
    __builtin_amdgcn_tensor_load_to_lds(g0, g1, z4, z4, z8, 0);
}

// data_size=4B(2), pad_enable, pad_interval=32 DWORDs(4), pad_amount=4 DWORDs(3)
#define TDM_FLAGS_AS ((2 << 16) | (1 << 20) | (4 << 22) | (3 << 25))
// data_size=2B(1), pad_enable, pad_interval=16 DWORDs(3), pad_amount=4 DWORDs(3)
#define TDM_FLAGS_BS ((1 << 16) | (1 << 20) | (3 << 22) | (3 << 25))

// ---------------- Kernel 3: main WMMA GEMM + tanh + beta reduction -----------------------
// One block per s. 512 threads = 16 waves in 4(M) x 4(N) arrangement.
// TDM stages enc (f32) and W2 (bf16) tiles into padded LDS each K-chunk;
// waves consume via ds_load_b128 -> v_wmma_f32_16x16x32_bf16.

#define AS_STRIDE 36   // f32 elems per A row in LDS (32 + 4 pad), set by TDM pad fields
#define BS_STRIDE 40   // bf16 elems per B row in LDS (32 + 8 pad), set by TDM pad fields

__device__ inline v8bf cvt8(float4 a, float4 b) {
    v8bf r;
    r[0] = (__bf16)a.x; r[1] = (__bf16)a.y; r[2] = (__bf16)a.z; r[3] = (__bf16)a.w;
    r[4] = (__bf16)b.x; r[5] = (__bf16)b.y; r[6] = (__bf16)b.z; r[7] = (__bf16)b.w;
    return r;
}

__global__ void __launch_bounds__(512)
attn_energy_score_kernel(const float* __restrict__ enc,
                         const __bf16* __restrict__ w2,
                         const float* __restrict__ hterm,
                         const float* __restrict__ beta,
                         float* __restrict__ scores) {
    __shared__ __align__(16) float  As[64 * AS_STRIDE];        // 9216 B
    __shared__ __align__(16) __bf16 Bs[512 * BS_STRIDE];       // 40960 B
    __shared__ float sc[64];

    const int s    = blockIdx.x;
    const int tid  = threadIdx.x;
    const int lane = tid & 31;
    const int wave = tid >> 5;
    const int wm   = wave & 3;    // M slab: rows [wm*16, wm*16+16)
    const int wn   = wave >> 2;   // N slab: cols [wn*128, wn*128+128)

    if (tid < 64) sc[tid] = 0.0f;

    const unsigned ldsAs = (unsigned)(uintptr_t)(void*)As;
    const unsigned ldsBs = (unsigned)(uintptr_t)(void*)Bs;

    // A-frag addressing (16-bit A 16x32 layout): M = lane%16, K halves at (lane/16)*8 and +16
    const int am   = lane & 15;
    const int akb  = (lane >> 4) << 3;     // 0 or 8
    // B-frag addressing: N = lane%16, K halves at (lane/16)*8 and +16
    const int bkb  = (lane >> 4) << 3;     // 0 or 8

    v8f acc[8];
#pragma unroll
    for (int nt = 0; nt < 8; ++nt) acc[nt] = (v8f){0,0,0,0,0,0,0,0};

    for (int kc = 0; kc < 512; kc += 32) {
        __syncthreads();   // previous chunk's compute done before TDM overwrites LDS

        if (wave == 0) {
            // enc tile: 64 rows x 32 f32, row stride 512 f32
            tdm_load_2d(ldsAs, enc + (size_t)s * 64 * 512 + kc,
                        TDM_FLAGS_AS, /*tile_d0=*/32, /*tile_d1=*/64,
                        /*tensor_d0=*/512, /*stride0=*/512);
            // W2 tile: 512 rows x 32 bf16, row stride 512 bf16
            tdm_load_2d(ldsBs, w2 + kc,
                        TDM_FLAGS_BS, /*tile_d0=*/32, /*tile_d1=*/512,
                        /*tensor_d0=*/512, /*stride0=*/512);
            __builtin_amdgcn_s_wait_tensorcnt(0);
        }
        __syncthreads();   // LDS tiles visible to all waves

        // ---- build A fragment (convert f32 -> bf16)
        const float* arow_p = &As[(wm * 16 + am) * AS_STRIDE];
        float4 x0 = *(const float4*)(arow_p + akb);
        float4 x1 = *(const float4*)(arow_p + akb + 4);
        float4 x2 = *(const float4*)(arow_p + akb + 16);
        float4 x3 = *(const float4*)(arow_p + akb + 20);
        v16bf af = __builtin_shufflevector(cvt8(x0, x1), cvt8(x2, x3),
                                           0,1,2,3,4,5,6,7,8,9,10,11,12,13,14,15);

        // ---- 8 WMMAs over the wave's 128 N columns
#pragma unroll
        for (int nt = 0; nt < 8; ++nt) {
            const int n = wn * 128 + nt * 16 + (lane & 15);
            v8bf b0 = *(const v8bf*)&Bs[n * BS_STRIDE + bkb];
            v8bf b1 = *(const v8bf*)&Bs[n * BS_STRIDE + bkb + 16];
            v16bf bf = __builtin_shufflevector(b0, b1,
                                               0,1,2,3,4,5,6,7,8,9,10,11,12,13,14,15);
            acc[nt] = __builtin_amdgcn_wmma_f32_16x16x32_bf16(
                false, af, false, bf, (short)0, acc[nt], false, false);
        }
    }

    // ---- epilogue: tanh + beta dot, reduce per row
    float rowpart[8] = {0,0,0,0,0,0,0,0};
    const int mbase = wm * 16 + ((lane >> 4) << 3);   // C layout: M = j + 8*(lane/16)
#pragma unroll
    for (int nt = 0; nt < 8; ++nt) {
        const int n = wn * 128 + nt * 16 + (lane & 15);
        const float bv = beta[n];
#pragma unroll
        for (int j = 0; j < 8; ++j) {
            const int b = mbase + j;
            const float e = tanhf(acc[nt][j] + hterm[(size_t)b * 512 + n]);
            rowpart[j] = fmaf(bv, e, rowpart[j]);
        }
    }
#pragma unroll
    for (int j = 0; j < 8; ++j) atomicAdd(&sc[mbase + j], rowpart[j]);

    __syncthreads();
    if (tid < 64) scores[(size_t)tid * S_LEN + s] = sc[tid];   // scores[b][s]
}

// ---------------- Kernel 4: softmax over S per batch column ------------------------------
__global__ void softmax_kernel(const float* __restrict__ scores, float* __restrict__ out) {
    const int b = blockIdx.x;       // 64 blocks
    const int t = threadIdx.x;      // 256 threads
    __shared__ float red[256];

    float v[4];
#pragma unroll
    for (int i = 0; i < 4; ++i) v[i] = scores[(size_t)b * S_LEN + t + i * 256];

    float mx = fmaxf(fmaxf(v[0], v[1]), fmaxf(v[2], v[3]));
    red[t] = mx;
    __syncthreads();
    for (int off = 128; off > 0; off >>= 1) {
        if (t < off) red[t] = fmaxf(red[t], red[t + off]);
        __syncthreads();
    }
    mx = red[0];
    __syncthreads();

    float e[4], sum = 0.0f;
#pragma unroll
    for (int i = 0; i < 4; ++i) { e[i] = expf(v[i] - mx); sum += e[i]; }
    red[t] = sum;
    __syncthreads();
    for (int off = 128; off > 0; off >>= 1) {
        if (t < off) red[t] += red[t + off];
        __syncthreads();
    }
    const float inv = 1.0f / red[0];
#pragma unroll
    for (int i = 0; i < 4; ++i) {
        const int s = t + i * 256;
        out[(size_t)s * B_SZ + b] = e[i] * inv;   // [S,1,B]
    }
}

// ---------------- Launch ----------------------------------------------------------------
extern "C" void kernel_launch(void* const* d_in, const int* in_sizes, int n_in,
                              void* d_out, int out_size, void* d_ws, size_t ws_size,
                              hipStream_t stream) {
    const float* hidden = (const float*)d_in[0];   // [1,64,512]
    const float* enc    = (const float*)d_in[1];   // [1024,64,512]
    const float* W_attn = (const float*)d_in[2];   // [512,1024]
    const float* b_attn = (const float*)d_in[3];   // [512]
    const float* beta   = (const float*)d_in[4];   // [512,1]
    float* out = (float*)d_out;                    // [1024,1,64]

    char* ws = (char*)d_ws;
    __bf16* w2     = (__bf16*)(ws);                        // 512*512*2 = 524288 B
    float*  hterm  = (float*)(ws + 524288);                // 64*512*4  = 131072 B
    float*  scores = (float*)(ws + 524288 + 131072);       // 64*1024*4 = 262144 B

    w2_convert_kernel<<<(512 * 512) / 256, 256, 0, stream>>>(W_attn, w2);
    hterm_kernel<<<(64 * 512) / 256, 256, 0, stream>>>(hidden, W_attn, b_attn, hterm);
    attn_energy_score_kernel<<<S_LEN, 512, 0, stream>>>(enc, w2, hterm, beta, scores);
    softmax_kernel<<<B_SZ, 256, 0, stream>>>(scores, out);
}
